// RecurrentGCN_3770981286463
// MI455X (gfx1250) — matrix operand order repaired and checked by reading, not compiled, not run on hardware
//
#include <hip/hip_runtime.h>
#include <hip/hip_bf16.h>
#include <math.h>

#define N_NODES 50000
#define N_EDGES 600000
#define D_FEAT  128

typedef __attribute__((ext_vector_type(16))) _Float16 v16h;
typedef __attribute__((ext_vector_type(4)))  _Float16 v4h;
typedef __attribute__((ext_vector_type(8)))  float    v8f;

__device__ __forceinline__ void atom_add_f32(float* p, float v) {
    __hip_atomic_fetch_add(p, v, __ATOMIC_RELAXED, __HIP_MEMORY_SCOPE_AGENT);
}

// Packed B-fragment position for element (k, j) of a 128x128 B matrix:
// fragment (kt = k/32, nt = j/16), lane = (j&15) + 16*((k%32)/8 & 1),
// element = (k%32)&7 + 8*(k%32 >= 16). Each lane's 16 halves are contiguous (32 B).
__device__ __forceinline__ int pack_pos(int k, int j) {
    int kt   = k >> 5;
    int kin  = k & 31;
    int lane = (j & 15) + ((kin & 8) ? 16 : 0);
    int nt   = j >> 4;
    return ((((kt << 3) + nt) << 5) + lane) * 16 + ((kin & 7) | ((kin & 16) >> 1));
}

// ---------------------------------------------------------------------------
// 1) GRU step on W0: W = (1-z)*n + z*W0. Stored as f16 directly in the
//    pre-swizzled WMMA B-fragment layout (k = row of W = reduction dim).
// ---------------------------------------------------------------------------
__global__ void gru_kernel(const float* __restrict__ W0,
                           const float* __restrict__ w_ih,
                           const float* __restrict__ w_hh,
                           const float* __restrict__ b_ih,
                           const float* __restrict__ b_hh,
                           _Float16* __restrict__ Wpack) {
    int idx = blockIdx.x * 256 + threadIdx.x;   // 0..16383
    int i = idx >> 7;                           // row of W (k dim)
    int j = idx & 127;                          // col of W (n dim)
    const float* w0r = W0 + (size_t)i * D_FEAT;

    float g[6];
#pragma unroll
    for (int t = 0; t < 3; ++t) {
        const float* wi = w_ih + (size_t)(t * 128 + j) * D_FEAT;
        const float* wh = w_hh + (size_t)(t * 128 + j) * D_FEAT;
        float si = 0.f, sh = 0.f;
        for (int d = 0; d < D_FEAT; ++d) {
            float w0v = w0r[d];
            si += w0v * wi[d];
            sh += w0v * wh[d];
        }
        g[t]     = si + b_ih[t * 128 + j];
        g[3 + t] = sh + b_hh[t * 128 + j];
    }
    float r = 1.f / (1.f + expf(-(g[0] + g[3])));
    float z = 1.f / (1.f + expf(-(g[1] + g[4])));
    float n = tanhf(g[2] + r * g[5]);
    float w = (1.f - z) * n + z * w0r[j];
    Wpack[pack_pos(i, j)] = (_Float16)w;
}

// ---------------------------------------------------------------------------
// 1b) Repack W_lin^T into the same fragment layout: B[k][j] = W_lin[j][k].
// ---------------------------------------------------------------------------
__global__ void repack_wlin_kernel(const float* __restrict__ W_lin,
                                   _Float16* __restrict__ WLpack) {
    int idx = blockIdx.x * 256 + threadIdx.x;   // 0..16383
    int j = idx >> 7;                           // output feature (n dim)
    int k = idx & 127;                          // input feature (k dim)
    WLpack[pack_pos(k, j)] = (_Float16)W_lin[(size_t)j * D_FEAT + k];
}

// ---------------------------------------------------------------------------
// 2) Degree / normalization
// ---------------------------------------------------------------------------
__global__ void deg_init_kernel(float* __restrict__ deg, int n) {
    int i = blockIdx.x * 256 + threadIdx.x;
    if (i < n) deg[i] = 1.0f;                    // self-loop weight
}

__global__ void deg_edge_kernel(const int* __restrict__ ei,
                                const float* __restrict__ ew,
                                float* __restrict__ deg, int e_cnt) {
    int e = blockIdx.x * 256 + threadIdx.x;
    if (e < e_cnt) atom_add_f32(&deg[ei[e_cnt + e]], ew[e]);   // col = target
}

__global__ void dinv_kernel(float* __restrict__ deg, int n) {
    int i = blockIdx.x * 256 + threadIdx.x;
    if (i < n) deg[i] = rsqrtf(deg[i]);          // deg >= 1 always
}

// ---------------------------------------------------------------------------
// 3) xw = x @ W  -> stored as f16 (halves scatter-gather traffic).
//    One wave per 16-row tile, full 128-col output. Per k-step: one A frag,
//    batch-load all 8 B frags, then 8 back-to-back WMMAs.
//    grid = 625 blocks x 160 threads (5 waves) = 3125 tiles exactly.
// ---------------------------------------------------------------------------
__global__ __launch_bounds__(160) void xw_wmma_kernel(
        const float* __restrict__ x,
        const v16h*  __restrict__ Wp,
        _Float16* __restrict__ xwh) {
    const int lane  = threadIdx.x & 31;
    const int wave  = threadIdx.x >> 5;          // 0..4
    const int row0  = (blockIdx.x * 5 + wave) * 16;
    const int r     = lane & 15;
    const int hi    = lane >> 4;                 // 0 or 1
    const int halfA = hi * 8;

    const float* xrow = x + (size_t)(row0 + r) * D_FEAT;
    v8f c[8];
#pragma unroll
    for (int nt = 0; nt < 8; ++nt) c[nt] = (v8f){};

#pragma unroll
    for (int kt = 0; kt < 4; ++kt) {
        const int k0 = kt * 32;
        v16h a;
#pragma unroll
        for (int i = 0; i < 8; ++i) {
            a[i]     = (_Float16)xrow[k0 + halfA + i];
            a[8 + i] = (_Float16)xrow[k0 + 16 + halfA + i];
        }
        v16h b[8];
#pragma unroll
        for (int nt = 0; nt < 8; ++nt)
            b[nt] = Wp[((kt << 3) + nt) * 32 + lane];
#pragma unroll
        for (int nt = 0; nt < 8; ++nt)
            c[nt] = __builtin_amdgcn_wmma_f32_16x16x32_f16(
                        false, a, false, b[nt], (short)0, c[nt], false, false);
    }
#pragma unroll
    for (int nt = 0; nt < 8; ++nt) {
        _Float16* orow = xwh + (size_t)(row0 + hi * 8) * D_FEAT + nt * 16 + r;
#pragma unroll
        for (int v = 0; v < 8; ++v) orow[(size_t)v * D_FEAT] = (_Float16)c[nt][v];
    }
}

// ---------------------------------------------------------------------------
// 4) h init with self-loop term: h[i] = dinv[i]^2 * xw[i]  (f16 in, f32 out)
// ---------------------------------------------------------------------------
__global__ void selfloop_kernel(const _Float16* __restrict__ xwh,
                                const float* __restrict__ dinv,
                                float* __restrict__ h) {
    size_t f4   = (size_t)blockIdx.x * 256 + threadIdx.x;  // N*128/4 total
    size_t node = f4 >> 5;                                 // 32 quads per row
    float  s    = dinv[node]; s *= s;
    v4h    v    = ((const v4h*)xwh)[f4];
    float4 o    = { (float)v[0] * s, (float)v[1] * s,
                    (float)v[2] * s, (float)v[3] * s };
    ((float4*)h)[f4] = o;
}

// ---------------------------------------------------------------------------
// 5) Edge scatter: h[dst] += dinv[src]*ew*dinv[dst] * xw[src]
//    1 wave per edge, 4 halves gathered per lane, hardware f32 atomics.
// ---------------------------------------------------------------------------
__global__ void scatter_kernel(const int* __restrict__ ei,
                               const float* __restrict__ ew,
                               const float* __restrict__ dinv,
                               const _Float16* __restrict__ xwh,
                               float* __restrict__ h) {
    int e    = (int)((blockIdx.x * 256 + threadIdx.x) >> 5);  // edge id
    int lane = threadIdx.x & 31;
    int src  = ei[e];
    int dst  = ei[N_EDGES + e];
    float norm = dinv[src] * ew[e] * dinv[dst];

    v4h v = ((const v4h*)(xwh + (size_t)src * D_FEAT))[lane];
    float* hp = h + (size_t)dst * D_FEAT + lane * 4;
    atom_add_f32(hp + 0, (float)v[0] * norm);
    atom_add_f32(hp + 1, (float)v[1] * norm);
    atom_add_f32(hp + 2, (float)v[2] * norm);
    atom_add_f32(hp + 3, (float)v[3] * norm);
}

// ---------------------------------------------------------------------------
// 6) out = relu(h) @ W_lin^T + b_lin  (same scheme; bias folded into C)
// ---------------------------------------------------------------------------
__global__ __launch_bounds__(160) void out_wmma_kernel(
        const float* __restrict__ h,
        const v16h*  __restrict__ WLp,
        const float* __restrict__ b_lin,
        float* __restrict__ out) {
    const int lane  = threadIdx.x & 31;
    const int wave  = threadIdx.x >> 5;
    const int row0  = (blockIdx.x * 5 + wave) * 16;
    const int r     = lane & 15;
    const int hi    = lane >> 4;
    const int halfA = hi * 8;

    const float* hrow = h + (size_t)(row0 + r) * D_FEAT;
    v8f c[8];
#pragma unroll
    for (int nt = 0; nt < 8; ++nt) {
        float bias = b_lin[nt * 16 + r];
#pragma unroll
        for (int v = 0; v < 8; ++v) c[nt][v] = bias;
    }

#pragma unroll
    for (int kt = 0; kt < 4; ++kt) {
        const int k0 = kt * 32;
        v16h a;
#pragma unroll
        for (int i = 0; i < 8; ++i) {
            a[i]     = (_Float16)fmaxf(hrow[k0 + halfA + i], 0.f);
            a[8 + i] = (_Float16)fmaxf(hrow[k0 + 16 + halfA + i], 0.f);
        }
        v16h b[8];
#pragma unroll
        for (int nt = 0; nt < 8; ++nt)
            b[nt] = WLp[((kt << 3) + nt) * 32 + lane];
#pragma unroll
        for (int nt = 0; nt < 8; ++nt)
            c[nt] = __builtin_amdgcn_wmma_f32_16x16x32_f16(
                        false, a, false, b[nt], (short)0, c[nt], false, false);
    }
#pragma unroll
    for (int nt = 0; nt < 8; ++nt) {
        float* orow = out + (size_t)(row0 + hi * 8) * D_FEAT + nt * 16 + r;
#pragma unroll
        for (int v = 0; v < 8; ++v) orow[(size_t)v * D_FEAT] = c[nt][v];
    }
}

// ---------------------------------------------------------------------------
extern "C" void kernel_launch(void* const* d_in, const int* in_sizes, int n_in,
                              void* d_out, int out_size, void* d_ws, size_t ws_size,
                              hipStream_t stream) {
    (void)in_sizes; (void)n_in; (void)out_size; (void)ws_size;
    const float* x     = (const float*)d_in[0];
    const int*   ei    = (const int*)  d_in[1];
    const float* ew    = (const float*)d_in[2];
    const float* W0    = (const float*)d_in[3];
    const float* w_ih  = (const float*)d_in[4];
    const float* w_hh  = (const float*)d_in[5];
    const float* b_ih  = (const float*)d_in[6];
    const float* b_hh  = (const float*)d_in[7];
    const float* W_lin = (const float*)d_in[8];
    const float* b_lin = (const float*)d_in[9];
    float* out = (float*)d_out;

    char* ws = (char*)d_ws;
    _Float16* Wpack  = (_Float16*)ws;                    // 32 KB (packed B frags)
    _Float16* WLpack = (_Float16*)(ws + 32768);          // 32 KB
    float*    deg    = (float*)(ws + 65536);             // N*4 (dinv in place)
    _Float16* xwh    = (_Float16*)(ws + 65536 + 262144); // N*128*2 (f16)
    float*    h      = (float*)(ws + 65536 + 262144 + (size_t)N_NODES * D_FEAT * 2);

    gru_kernel        <<<64, 256, 0, stream>>>(W0, w_ih, w_hh, b_ih, b_hh, Wpack);
    repack_wlin_kernel<<<64, 256, 0, stream>>>(W_lin, WLpack);
    deg_init_kernel   <<<(N_NODES + 255) / 256, 256, 0, stream>>>(deg, N_NODES);
    deg_edge_kernel   <<<(N_EDGES + 255) / 256, 256, 0, stream>>>(ei, ew, deg, N_EDGES);
    dinv_kernel       <<<(N_NODES + 255) / 256, 256, 0, stream>>>(deg, N_NODES);
    xw_wmma_kernel    <<<N_NODES / 16 / 5, 160, 0, stream>>>(x, (const v16h*)Wpack, xwh);
    selfloop_kernel   <<<(N_NODES * (D_FEAT / 4)) / 256, 256, 0, stream>>>(xwh, deg, h);
    scatter_kernel    <<<N_EDGES / 8, 256, 0, stream>>>(ei, ew, deg, xwh, h);
    out_wmma_kernel   <<<N_NODES / 16 / 5, 160, 0, stream>>>(h, (const v16h*)WLpack, b_lin, out);
}
